// Encoder_59820304498737
// MI455X (gfx1250) — matrix operand order repaired and checked by reading, not compiled, not run on hardware
//
#include <hip/hip_runtime.h>

// ---------------- CDNA5 WMMA types ----------------
typedef __attribute__((ext_vector_type(16))) __bf16    v16bf;
typedef __attribute__((ext_vector_type(8)))  float     v8f;
typedef __attribute__((ext_vector_type(8)))  unsigned  v8u;

__device__ __forceinline__ v8f zero_v8f() {
    v8f z = {0.f,0.f,0.f,0.f,0.f,0.f,0.f,0.f};
    return z;
}

__device__ __forceinline__ unsigned short f2bf(float f) {
    unsigned u = __builtin_bit_cast(unsigned, f);
    unsigned r = u + 0x7FFFu + ((u >> 16) & 1u);   // round-to-nearest-even
    return (unsigned short)(r >> 16);
}

// Build a 16xbf16 fragment from two 16B-aligned runs (LDS or global).
__device__ __forceinline__ v16bf ld_frag(const unsigned short* p0, const unsigned short* p1) {
    uint4 a = *reinterpret_cast<const uint4*>(p0);
    uint4 b = *reinterpret_cast<const uint4*>(p1);
    v8u u;
    u[0]=a.x; u[1]=a.y; u[2]=a.z; u[3]=a.w;
    u[4]=b.x; u[5]=b.y; u[6]=b.z; u[7]=b.w;
    return __builtin_bit_cast(v16bf, u);
}

#define WMMA_BF16(a, b, c) \
    __builtin_amdgcn_wmma_f32_16x16x32_bf16(false, (a), false, (b), (short)0, (c), false, false)

// ---------------- CDNA5 async global->LDS copy (ASYNCcnt) ----------------
// ISA §15.18.3 opcode 98: GLOBAL_LOAD_ASYNC_TO_LDS_B128 (GV mode).
__device__ __forceinline__ void async_ld16(unsigned ldsoff, const unsigned short* g) {
    asm volatile("global_load_async_to_lds_b128 %0, %1, off"
                 :: "v"(ldsoff), "v"((unsigned long long)(size_t)g)
                 : "memory");
}
__device__ __forceinline__ void wait_async8() { asm volatile("s_wait_asynccnt 0x8" ::: "memory"); }
__device__ __forceinline__ void wait_async0() { asm volatile("s_wait_asynccnt 0x0" ::: "memory"); }

// ============================================================
// GEMM: Cf/Cb[M,N] = act(A[M,K](bf16) * Bt[N,K]^T(bf16) + bias)
// 128x128 block tile, 128 threads = 4 waves (2x2), 64x64 per wave,
// K-step 32, double-buffered async global->LDS staging, peeled loop.
// 16 WMMA per 16 ds_load_b128 per wave per K-step.
// ============================================================
__global__ __launch_bounds__(128) void gemm_async_wmma_kernel(
    const unsigned short* __restrict__ A,    // [M,K] bf16 activations
    const unsigned short* __restrict__ Bt,   // [N,K] bf16 pre-transposed weight
    const float* __restrict__ bias,
    float* __restrict__ Cf,                  // optional fp32 out
    unsigned short* __restrict__ Cb,         // optional bf16 out
    int M, int N, int K, int relu)
{
    __shared__ __align__(16) unsigned short As[2][128][40];  // [buf][m][k], +8 pad
    __shared__ __align__(16) unsigned short Bs[2][128][40];  // [buf][n][k], +8 pad

    const int tid  = threadIdx.x;
    const int m0   = blockIdx.x * 128;
    const int n0   = blockIdx.y * 128;
    const int wid  = tid >> 5;
    const int lane = tid & 31;
    const int half = lane >> 4;
    const int lr   = lane & 15;
    const int wm   = wid & 1;   // 2 waves along M (64 rows each)
    const int wn   = wid >> 1;  // 2 waves along N (64 cols each)

    const unsigned aBase = (unsigned)(size_t)&As[0][0][0];
    const unsigned bBase = (unsigned)(size_t)&Bs[0][0][0];
    const unsigned BUFB  = 128u * 40u * 2u;  // bytes per LDS buffer

    v8f acc[4][4];
    #pragma unroll
    for (int ms = 0; ms < 4; ++ms)
        #pragma unroll
        for (int ns = 0; ns < 4; ++ns)
            acc[ms][ns] = zero_v8f();

    // 512 16B chunks per matrix per tile; 4 per thread per matrix.
    // idx = tid + j*128: lanes 0..3 -> row0 c0..3, lanes 4..7 -> row1, ...
    auto issue = [&](int kt, int buf) {
        const int k0 = kt * 32;
        #pragma unroll
        for (int j = 0; j < 4; ++j) {
            int idx = tid + j * 128;
            int row = idx >> 2;
            int c   = idx & 3;
            unsigned loff = (unsigned)(buf * BUFB + row * 80 + c * 16);
            async_ld16(aBase + loff, A  + (size_t)(m0 + row) * K + k0 + c * 8);
            async_ld16(bBase + loff, Bt + (size_t)(n0 + row) * K + k0 + c * 8);
        }
    };

    auto compute = [&](int cur) {
        v16bf af[4], bfr[4];
        #pragma unroll
        for (int ms = 0; ms < 4; ++ms) {
            int ar = wm * 64 + ms * 16 + lr;
            af[ms] = ld_frag(&As[cur][ar][half * 8], &As[cur][ar][16 + half * 8]);
        }
        #pragma unroll
        for (int ns = 0; ns < 4; ++ns) {
            int br = wn * 64 + ns * 16 + lr;
            bfr[ns] = ld_frag(&Bs[cur][br][half * 8], &Bs[cur][br][16 + half * 8]);
        }
        #pragma unroll
        for (int ms = 0; ms < 4; ++ms)
            #pragma unroll
            for (int ns = 0; ns < 4; ++ns)
                acc[ms][ns] = WMMA_BF16(af[ms], bfr[ns], acc[ms][ns]);
    };

    const int NT = K >> 5;  // >= 2 for all shapes here
    issue(0, 0);
    // steady state: straight-line issue/wait/compute
    for (int it = 0; it < NT - 1; ++it) {
        const int cur = it & 1;
        issue(it + 1, cur ^ 1);
        wait_async8();          // previous tile complete; next 8/thread in flight
        __syncthreads();
        compute(cur);
        __syncthreads();        // protect buffer reuse before next issue
    }
    // peeled last iteration
    wait_async0();
    __syncthreads();
    compute((NT - 1) & 1);

    // ---- epilogue ----
    for (int ns = 0; ns < 4; ++ns) {
        int col = n0 + wn * 64 + ns * 16 + lr;
        float bv = bias[col];
        for (int ms = 0; ms < 4; ++ms) {
            #pragma unroll
            for (int r = 0; r < 8; ++r) {
                int row = m0 + wm * 64 + ms * 16 + r + half * 8;
                float v = acc[ms][ns][r] + bv;
                if (relu) v = fmaxf(v, 0.f);
                if (Cf) Cf[(size_t)row * N + col] = v;
                if (Cb) Cb[(size_t)row * N + col] = f2bf(v);
            }
        }
    }
}

// ============================================================
// Attention: one wave per (16-query tile, head, batch), bf16 in/out.
// ============================================================
#define SLEN 512
__global__ __launch_bounds__(32) void attn_wmma_kernel(
    const unsigned short* __restrict__ Q, const unsigned short* __restrict__ K,
    const unsigned short* __restrict__ V, unsigned short* __restrict__ O)
{
    __shared__ __align__(16) float          Es[16][SLEN];        // 32 KB
    __shared__ __align__(16) unsigned short Ps[16][SLEN + 8];    // bf16 probs
    __shared__ __align__(16) unsigned short Vt[64][40];          // V^T tile
    __shared__ float rsum[16];

    const int qt   = blockIdx.x;
    const int h    = blockIdx.y;
    const int b    = blockIdx.z;
    const int lane = threadIdx.x;
    const int half = lane >> 4;
    const int lr   = lane & 15;
    const float scale = 0.044194173824159216f;  // 1/sqrt(512)

    const unsigned short* qrow = Q + ((size_t)(b * SLEN + qt * 16 + lr)) * 512 + h * 64;
    v16bf qa0 = ld_frag(qrow + half * 8,      qrow + 16 + half * 8);
    v16bf qa1 = ld_frag(qrow + 32 + half * 8, qrow + 48 + half * 8);

    // ---- pass 1: energies into LDS ----
    for (int kt = 0; kt < SLEN / 16; ++kt) {
        const unsigned short* krow = K + ((size_t)(b * SLEN + kt * 16 + lr)) * 512 + h * 64;
        v16bf kb0 = ld_frag(krow + half * 8,      krow + 16 + half * 8);
        v16bf kb1 = ld_frag(krow + 32 + half * 8, krow + 48 + half * 8);
        v8f e = zero_v8f();
        e = WMMA_BF16(qa0, kb0, e);
        e = WMMA_BF16(qa1, kb1, e);
        #pragma unroll
        for (int r = 0; r < 8; ++r)
            Es[r + half * 8][kt * 16 + lr] = e[r] * scale;
    }
    __syncthreads();

    // ---- softmax: lane handles row lr, keys [half*256, +256) ----
    {
        const int row = lr;
        float m = -1e30f;
        for (int j = 0; j < 256; ++j)
            m = fmaxf(m, Es[row][half * 256 + j]);
        m = fmaxf(m, __shfl_xor(m, 16, 32));
        float s = 0.f;
        for (int j = 0; j < 256; ++j) {
            float e = __expf(Es[row][half * 256 + j] - m);
            Ps[row][half * 256 + j] = f2bf(e);
            s += e;
        }
        s += __shfl_xor(s, 16, 32);
        if (half == 0) rsum[row] = 1.0f / s;
    }
    __syncthreads();

    // ---- pass 2: O = P V ----
    v8f oacc[4];
    for (int ns = 0; ns < 4; ++ns) oacc[ns] = zero_v8f();

    for (int ks = 0; ks < 16; ++ks) {
        __syncthreads();
        {   // stage V^T tile: 32 keys x 64 dh -> Vt[dh][key]
            int key = ks * 32 + lane;
            const unsigned short* vrow = V + ((size_t)(b * SLEN + key)) * 512 + h * 64;
            #pragma unroll
            for (int j8 = 0; j8 < 8; ++j8) {
                uint4 u = ((const uint4*)vrow)[j8];
                Vt[j8 * 8 + 0][lane] = (unsigned short)u.x;
                Vt[j8 * 8 + 1][lane] = (unsigned short)(u.x >> 16);
                Vt[j8 * 8 + 2][lane] = (unsigned short)u.y;
                Vt[j8 * 8 + 3][lane] = (unsigned short)(u.y >> 16);
                Vt[j8 * 8 + 4][lane] = (unsigned short)u.z;
                Vt[j8 * 8 + 5][lane] = (unsigned short)(u.z >> 16);
                Vt[j8 * 8 + 6][lane] = (unsigned short)u.w;
                Vt[j8 * 8 + 7][lane] = (unsigned short)(u.w >> 16);
            }
        }
        __syncthreads();
        v16bf pa = ld_frag(&Ps[lr][ks * 32 + half * 8],
                           &Ps[lr][ks * 32 + 16 + half * 8]);
        #pragma unroll
        for (int ns = 0; ns < 4; ++ns) {
            v16bf vb = ld_frag(&Vt[ns * 16 + lr][half * 8],
                               &Vt[ns * 16 + lr][16 + half * 8]);
            oacc[ns] = WMMA_BF16(pa, vb, oacc[ns]);
        }
    }

    // ---- epilogue: /rowsum, write bf16 ctx in [b,s,h*64+col] ----
    for (int ns = 0; ns < 4; ++ns) {
        #pragma unroll
        for (int r = 0; r < 8; ++r) {
            int row = r + half * 8;
            O[((size_t)(b * SLEN + qt * 16 + row)) * 512 + h * 64 + ns * 16 + lr]
                = f2bf(oacc[ns][r] * rsum[row]);
        }
    }
}

// ============================================================
// x = x*sqrt(512) + PE(s,d); writes fp32 and bf16 copies
// ============================================================
__global__ __launch_bounds__(256) void posenc_kernel(float* __restrict__ x,
                                                     unsigned short* __restrict__ xb)
{
    const int token = blockIdx.x;
    const int t     = threadIdx.x;
    const int s     = token & 511;
    float* p = x + (size_t)token * 512;
    #pragma unroll
    for (int j = 0; j < 2; ++j) {
        int d  = t + j * 256;
        int i2 = d & ~1;
        float freq = __expf(-0.0179889460f * (float)i2);  // ln(10000)/512
        float ang  = (float)s * freq;
        float pe   = (d & 1) ? __cosf(ang) : __sinf(ang);
        float v    = p[d] * 22.627416997969522f + pe;     // sqrt(512)
        p[d] = v;
        xb[(size_t)token * 512 + d] = f2bf(v);
    }
}

// ============================================================
// out = LayerNorm(x + r)*scale + bias; fp32 + bf16 outputs
// ============================================================
__global__ __launch_bounds__(256) void add_ln_kernel(
    const float* __restrict__ x, const float* __restrict__ r,
    const float* __restrict__ s, const float* __restrict__ bgain,
    float* __restrict__ outf, unsigned short* __restrict__ outb)
{
    __shared__ float red[256];
    const int token = blockIdx.x;
    const int t     = threadIdx.x;
    const float* xr = x + (size_t)token * 512;
    const float* rr = r + (size_t)token * 512;

    float v0 = xr[t] + rr[t];
    float v1 = xr[t + 256] + rr[t + 256];

    red[t] = v0 + v1;
    __syncthreads();
    for (int o = 128; o > 0; o >>= 1) {
        if (t < o) red[t] += red[t + o];
        __syncthreads();
    }
    float mu = red[0] * (1.0f / 512.0f);
    __syncthreads();

    float d0 = v0 - mu, d1 = v1 - mu;
    red[t] = d0 * d0 + d1 * d1;
    __syncthreads();
    for (int o = 128; o > 0; o >>= 1) {
        if (t < o) red[t] += red[t + o];
        __syncthreads();
    }
    float rstd = rsqrtf(red[0] * (1.0f / 512.0f) + 1e-5f);

    float y0 = d0 * rstd * s[t]       + bgain[t];
    float y1 = d1 * rstd * s[t + 256] + bgain[t + 256];
    outf[(size_t)token * 512 + t]       = y0;
    outf[(size_t)token * 512 + t + 256] = y1;
    outb[(size_t)token * 512 + t]       = f2bf(y0);
    outb[(size_t)token * 512 + t + 256] = f2bf(y1);
}

// ============================================================
// Weight prep: W[l][K][N] fp32 -> Wt[l][N][K] bf16 (tiled transpose)
// ============================================================
__global__ __launch_bounds__(256) void transpose_cvt_kernel(
    const float* __restrict__ W, unsigned short* __restrict__ Wt, int K, int N)
{
    __shared__ float tile[32][33];
    const int l  = blockIdx.z;
    const float* w = W + (size_t)l * K * N;
    unsigned short* wt = Wt + (size_t)l * K * N;
    const int bn = blockIdx.x * 32;
    const int bk = blockIdx.y * 32;
    const int tx = threadIdx.x & 31;
    const int ty = threadIdx.x >> 5;   // 0..7
    #pragma unroll
    for (int rr = 0; rr < 32; rr += 8)
        tile[ty + rr][tx] = w[(size_t)(bk + ty + rr) * N + bn + tx];
    __syncthreads();
    #pragma unroll
    for (int rr = 0; rr < 32; rr += 8)
        wt[(size_t)(bn + ty + rr) * K + bk + tx] = f2bf(tile[tx][ty + rr]);
}

__global__ __launch_bounds__(256) void cvt_bf16_kernel(
    const float* __restrict__ in, unsigned short* __restrict__ out)
{
    int i = blockIdx.x * 256 + threadIdx.x;
    out[i] = f2bf(in[i]);
}

// ============================================================
// Host-side orchestration
// ============================================================
extern "C" void kernel_launch(void* const* d_in, const int* in_sizes, int n_in,
                              void* d_out, int out_size, void* d_ws, size_t ws_size,
                              hipStream_t stream)
{
    (void)in_sizes; (void)n_in; (void)out_size; (void)ws_size;
    typedef unsigned short u16;

    const float* source = (const float*)d_in[0];
    const float* fin_w1 = (const float*)d_in[2];
    const float* fin_b1 = (const float*)d_in[3];
    const float* fin_w2 = (const float*)d_in[4];
    const float* fin_b2 = (const float*)d_in[5];
    const float* wq = (const float*)d_in[6];   const float* bq = (const float*)d_in[7];
    const float* wk = (const float*)d_in[8];   const float* bk = (const float*)d_in[9];
    const float* wv = (const float*)d_in[10];  const float* bv = (const float*)d_in[11];
    const float* wo = (const float*)d_in[12];  const float* bo = (const float*)d_in[13];
    const float* n1s = (const float*)d_in[14]; const float* n1b = (const float*)d_in[15];
    const float* f1w = (const float*)d_in[16]; const float* f1b = (const float*)d_in[17];
    const float* f2w = (const float*)d_in[18]; const float* f2b = (const float*)d_in[19];
    const float* n2s = (const float*)d_in[20]; const float* n2b = (const float*)d_in[21];
    float* out = (float*)d_out;

    const int M = 4096;  // B*S
    // ---- workspace carve ----
    float* x  = (float*)d_ws;                  // 4096*512 fp32
    float* tb = x + (size_t)M * 512;           // 4096*512 fp32
    u16* u    = (u16*)(tb + (size_t)M * 512);
    u16* srcb = u;              u += (size_t)M * 64;     // source bf16
    u16* xb   = u;              u += (size_t)M * 512;
    u16* hb   = u;              u += (size_t)M * 2048;
    u16* qb   = u;              u += (size_t)M * 512;
    u16* kb   = u;              u += (size_t)M * 512;
    u16* vb   = u;              u += (size_t)M * 512;
    u16* cb   = u;              u += (size_t)M * 512;
    u16* fw1t = u;              u += (size_t)64 * 2048;
    u16* fw2t = u;              u += (size_t)2048 * 512;
    u16* wqt  = u;              u += (size_t)6 * 512 * 512;
    u16* wkt  = u;              u += (size_t)6 * 512 * 512;
    u16* wvt  = u;              u += (size_t)6 * 512 * 512;
    u16* wot  = u;              u += (size_t)6 * 512 * 512;
    u16* f1t  = u;              u += (size_t)6 * 512 * 2048;
    u16* f2t  = u;              u += (size_t)6 * 2048 * 512;

    const dim3 blk(256);
    const dim3 gblk(128);      // GEMM: 4 waves
    const dim3 g512(32, 4);    // N=512
    const dim3 g2048(32, 16);  // N=2048

    // ---- weight prep: fp32 [K,N] -> bf16 [N,K] (L2-resident for the run) ----
    transpose_cvt_kernel<<<dim3(64, 2, 1),  blk, 0, stream>>>(fin_w1, fw1t, 64, 2048);
    transpose_cvt_kernel<<<dim3(16, 64, 1), blk, 0, stream>>>(fin_w2, fw2t, 2048, 512);
    transpose_cvt_kernel<<<dim3(16, 16, 6), blk, 0, stream>>>(wq,  wqt, 512, 512);
    transpose_cvt_kernel<<<dim3(16, 16, 6), blk, 0, stream>>>(wk,  wkt, 512, 512);
    transpose_cvt_kernel<<<dim3(16, 16, 6), blk, 0, stream>>>(wv,  wvt, 512, 512);
    transpose_cvt_kernel<<<dim3(16, 16, 6), blk, 0, stream>>>(wo,  wot, 512, 512);
    transpose_cvt_kernel<<<dim3(64, 16, 6), blk, 0, stream>>>(f1w, f1t, 512, 2048);
    transpose_cvt_kernel<<<dim3(16, 64, 6), blk, 0, stream>>>(f2w, f2t, 2048, 512);
    cvt_bf16_kernel<<<dim3(1024), blk, 0, stream>>>(source, srcb);

    // ---- input FFN + positional encoding ----
    gemm_async_wmma_kernel<<<g2048, gblk, 0, stream>>>(srcb, fw1t, fin_b1, nullptr, hb, M, 2048, 64, 1);
    gemm_async_wmma_kernel<<<g512,  gblk, 0, stream>>>(hb, fw2t, fin_b2, x, nullptr, M, 512, 2048, 0);
    posenc_kernel<<<dim3(4096), blk, 0, stream>>>(x, xb);

    for (int i = 0; i < 6; ++i) {
        const u16* wqi = wqt + (size_t)i * 512 * 512;
        const u16* wki = wkt + (size_t)i * 512 * 512;
        const u16* wvi = wvt + (size_t)i * 512 * 512;
        const u16* woi = wot + (size_t)i * 512 * 512;
        const u16* f1i = f1t + (size_t)i * 512 * 2048;
        const u16* f2i = f2t + (size_t)i * 2048 * 512;

        gemm_async_wmma_kernel<<<g512, gblk, 0, stream>>>(xb, wqi, bq + i * 512, nullptr, qb, M, 512, 512, 0);
        gemm_async_wmma_kernel<<<g512, gblk, 0, stream>>>(xb, wki, bk + i * 512, nullptr, kb, M, 512, 512, 0);
        gemm_async_wmma_kernel<<<g512, gblk, 0, stream>>>(xb, wvi, bv + i * 512, nullptr, vb, M, 512, 512, 0);

        attn_wmma_kernel<<<dim3(32, 8, 8), dim3(32), 0, stream>>>(qb, kb, vb, cb);

        gemm_async_wmma_kernel<<<g512, gblk, 0, stream>>>(cb, woi, bo + i * 512, tb, nullptr, M, 512, 512, 0);
        add_ln_kernel<<<dim3(4096), blk, 0, stream>>>(x, tb, n1s + i * 512, n1b + i * 512, x, xb);

        gemm_async_wmma_kernel<<<g2048, gblk, 0, stream>>>(xb, f1i, f1b + i * 2048, nullptr, hb, M, 2048, 512, 1);
        gemm_async_wmma_kernel<<<g512,  gblk, 0, stream>>>(hb, f2i, f2b + i * 512, tb, nullptr, M, 512, 2048, 0);
        add_ln_kernel<<<dim3(4096), blk, 0, stream>>>(x, tb, n2s + i * 512, n2b + i * 512,
                                                      (i == 5) ? out : x, xb);
    }
}
